// MultiheadAttention_13030930776796
// MI455X (gfx1250) — matrix operand order repaired and checked by reading, not compiled, not run on hardware
//
#include <hip/hip_runtime.h>
#include <cstdint>

#define DI static __device__ __forceinline__

typedef __attribute__((ext_vector_type(16))) __bf16 v16bf;
typedef __attribute__((ext_vector_type(8)))  float  v8f;
typedef unsigned int v4u __attribute__((ext_vector_type(4)));
typedef int          v8i __attribute__((ext_vector_type(8)));
typedef int          v4i __attribute__((ext_vector_type(4)));

constexpr int BB = 2, LL = 2048, DM = 1024, NH = 16, DK = 64, DV = 64;
constexpr float SCALE = 0.125f; // 1/sqrt(64)

#if defined(__HIP_DEVICE_COMPILE__)
  #if __has_builtin(__builtin_amdgcn_tensor_load_to_lds) && __has_builtin(__builtin_amdgcn_s_wait_tensorcnt)
    #define HAVE_TDM 1
  #else
    #define HAVE_TDM 0
  #endif
#else
  #define HAVE_TDM 0
#endif

// ---------------- WMMA wrapper ----------------
DI v8f wmma_bf16(v16bf a, v16bf b, v8f c) {
  // D = A(16x32 bf16) x B(32x16 bf16) + C(16x16 f32)
  return __builtin_amdgcn_wmma_f32_16x16x32_bf16(false, a, false, b,
                                                 (short)0, c, false, false);
}

// ---------------- fragment loaders (row-major bf16 memory) ----------------
// A-matrix 16x32 bf16 (MxK): lane<16 -> row M=lane, K={0..7,16..23};
// lane>=16 -> row M=lane-16, K={8..15,24..31}.
DI v16bf load_mat_A(const __bf16* base, int ld, int row0, int k0) {
  int lane = threadIdx.x & 31;
  const __bf16* p = base + (size_t)(row0 + (lane & 15)) * ld + k0 + ((lane & 16) ? 8 : 0);
  v16bf a;
#pragma unroll
  for (int i = 0; i < 8; ++i) { a[i] = p[i]; a[8 + i] = p[16 + i]; }
  return a;
}

// B-matrix 32x16 bf16 (KxN), memory holds Bt[n][k] (row-major, ld = K stride).
// lane<16 -> col N=lane, K={0..7,16..23}; lane>=16 -> col N=lane-16, K={8..15,24..31}.
DI v16bf load_mat_B(const __bf16* base, int ld, int n0, int k0) {
  int lane = threadIdx.x & 31;
  const __bf16* p = base + (size_t)(n0 + (lane & 15)) * ld + k0 + ((lane & 16) ? 8 : 0);
  v16bf b;
#pragma unroll
  for (int i = 0; i < 8; ++i) { b[i] = p[i]; b[8 + i] = p[16 + i]; }
  return b;
}

// ---------------- TDM staging: 64 rows x BK cols bf16 panel -> LDS ----------------
// Global pitch = DM elements; LDS pitch = BK+8 elems (TDM pad: 4 DWORDs per 64 DWORDs).
constexpr int BK = 128, LDW = BK + 8, NCH = DM / BK;

DI void tdm_stage_panel(const __bf16* src, unsigned lds_off) {
#if HAVE_TDM
  unsigned long long ga = (unsigned long long)(uintptr_t)src;
  v4u g0;
  g0[0] = 1u;                                   // count=1 valid descriptor
  g0[1] = lds_off;                              // lds_addr (bytes)
  g0[2] = (unsigned)ga;                         // global_addr[31:0]
  g0[3] = (unsigned)((ga >> 32) & 0x01FFFFFFull) | 0x80000000u; // addr[56:32] | type=2
  v8i g1;
  g1[0] = (int)((1u << 16) |                    // data_size = 2 bytes
                (1u << 20) |                    // pad_enable
                (5u << 22) |                    // pad_interval: 64 DWORDs (256B)
                (3u << 25));                    // pad_amount: 4 DWORDs (16B)
  g1[1] = (int)(((unsigned)DM & 0xFFFFu) << 16);        // tensor_dim0[15:0]
  g1[2] = (int)(((unsigned)DM >> 16) | (64u << 16));    // tensor_dim0[31:16] | tensor_dim1[15:0]
  g1[3] = (int)((unsigned)BK << 16);                    // tensor_dim1[31:16]=0 | tile_dim0
  g1[4] = 64;                                           // tile_dim1 | tile_dim2=0
  g1[5] = DM;                                           // tensor_dim0_stride[31:0]
  g1[6] = 0;                                            // stride0 hi | stride1 lo
  g1[7] = 0;
  v4i z4 = {0, 0, 0, 0};
#if __clang_major__ >= 23
  v8i z8 = {0, 0, 0, 0, 0, 0, 0, 0};
  __builtin_amdgcn_tensor_load_to_lds(g0, g1, z4, z4, z8, 0);
#else
  __builtin_amdgcn_tensor_load_to_lds(g0, g1, z4, z4, 0);
#endif
#else
  (void)src; (void)lds_off;
#endif
}

// ---------------- prep kernels ----------------
__global__ void cvt_bf16_kernel(const float* __restrict__ x, __bf16* __restrict__ y, int n) {
  int i = blockIdx.x * blockDim.x + threadIdx.x;
  if (i < n) y[i] = (__bf16)x[i];
}

// W[h][k][n] (f32) -> WT[h][n][k] (bf16)
__global__ void transpose_bf16_kernel(const float* __restrict__ W, __bf16* __restrict__ WT,
                                      int Kd, int Nd, int total) {
  int i = blockIdx.x * blockDim.x + threadIdx.x;
  if (i >= total) return;
  int h = i / (Kd * Nd);
  int r = i % (Kd * Nd);
  int k = r / Nd, n = r % Nd;
  WT[((size_t)h * Nd + n) * Kd + k] = (__bf16)W[i];
}

// ---------------- projection: Yp[b,h,:,:] = Xb[b] * W[h] + bias[h] ----------------
// 8 waves per block share (b,h); W panel double-buffered in LDS via TDM.
__global__ void proj_kernel(const __bf16* __restrict__ Xb, const __bf16* __restrict__ WT,
                            const float* __restrict__ bias, __bf16* __restrict__ Yp,
                            int transposed) {
  __shared__ __bf16 sW[2][64 * LDW];
  int wave = blockIdx.x * (blockDim.x >> 5) + (threadIdx.x >> 5);
  int qb = wave & 127;               // L/16 = 128 row tiles (8 consecutive per block)
  int h  = (wave >> 7) & (NH - 1);
  int b  = wave >> 11;
  const __bf16* A  = Xb + (size_t)b * LL * DM;
  const __bf16* Wh = WT + (size_t)h * DK * DM;

#if HAVE_TDM
  if (threadIdx.x < 32)
    tdm_stage_panel(Wh, (unsigned)reinterpret_cast<uintptr_t>(&sW[0][0]));
#endif

  v8f acc0 = {}, acc1 = {}, acc2 = {}, acc3 = {};
  for (int c = 0; c < NCH; ++c) {
    const int p = c & 1;
#if HAVE_TDM
    if (threadIdx.x < 32) {
      if (c + 1 < NCH) {
        tdm_stage_panel(Wh + (c + 1) * BK,
                        (unsigned)reinterpret_cast<uintptr_t>(&sW[p ^ 1][0]));
        __builtin_amdgcn_s_wait_tensorcnt(1);   // chunk c complete
      } else {
        __builtin_amdgcn_s_wait_tensorcnt(0);
      }
    }
#else
    for (int i = threadIdx.x; i < 64 * BK; i += 256) {
      int r = i >> 7, kk = i & (BK - 1);
      sW[p][r * LDW + kk] = Wh[(size_t)r * DM + c * BK + kk];
    }
#endif
    __syncthreads();
    const __bf16* sWp = &sW[p][0];
#pragma unroll
    for (int kk = 0; kk < BK; kk += 32) {
      v16bf a  = load_mat_A(A, DM, qb * 16, c * BK + kk);
      v16bf b0 = load_mat_B(sWp, LDW, 0,  kk);
      v16bf b1 = load_mat_B(sWp, LDW, 16, kk);
      v16bf b2 = load_mat_B(sWp, LDW, 32, kk);
      v16bf b3 = load_mat_B(sWp, LDW, 48, kk);
      acc0 = wmma_bf16(a, b0, acc0);
      acc1 = wmma_bf16(a, b1, acc1);
      acc2 = wmma_bf16(a, b2, acc2);
      acc3 = wmma_bf16(a, b3, acc3);
    }
    __syncthreads();
  }

  int lane = threadIdx.x & 31;
  int col = lane & 15;
  int rh = (lane & 16) ? 8 : 0;
  size_t head = (size_t)(b * NH + h);
  v8f accs[4] = {acc0, acc1, acc2, acc3};
#pragma unroll
  for (int nb = 0; nb < 4; ++nb) {
    int v = nb * 16 + col;
    float bv = bias[h * DK + v];
#pragma unroll
    for (int r = 0; r < 8; ++r) {
      int row = qb * 16 + rh + r;
      float val = accs[nb][r] + bv;
      if (!transposed) Yp[(head * LL + row) * DK + v] = (__bf16)val;
      else             Yp[(head * DK + v) * LL + row] = (__bf16)val;
    }
  }
}

// ---------------- column stats: softmax over the QUERY axis ----------------
__global__ void colstats_kernel(const __bf16* __restrict__ Qp, const __bf16* __restrict__ Kp,
                                float* __restrict__ mcol, float* __restrict__ rZ) {
  int wave = blockIdx.x * (blockDim.x >> 5) + (threadIdx.x >> 5);
  int sb = wave & 127;
  int h  = (wave >> 7) & (NH - 1);
  int b  = wave >> 11;
  const __bf16* Qh = Qp + (size_t)(b * NH + h) * LL * DK;
  const __bf16* Kh = Kp + (size_t)(b * NH + h) * LL * DK;

  v16bf ka0  = load_mat_A(Kh, DK, sb * 16, 0);   // fixed s-rows, loaded once
  v16bf ka32 = load_mat_A(Kh, DK, sb * 16, 32);

  float m[8], Z[8];
#pragma unroll
  for (int i = 0; i < 8; ++i) { m[i] = -3.0e38f; Z[i] = 0.0f; }

  for (int qb = 0; qb < LL; qb += 16) {
    v16bf qb0  = load_mat_B(Qh, DK, qb, 0);
    v16bf qb32 = load_mat_B(Qh, DK, qb, 32);
    v8f acc = {};
    acc = wmma_bf16(ka0,  qb0,  acc);
    acc = wmma_bf16(ka32, qb32, acc);
#pragma unroll
    for (int i = 0; i < 8; ++i) {
      float x = acc[i] * SCALE;
      float mn = fmaxf(m[i], x);
      Z[i] = Z[i] * __expf(m[i] - mn) + __expf(x - mn);
      m[i] = mn;
    }
  }
#pragma unroll
  for (int off = 1; off <= 8; off <<= 1) {
#pragma unroll
    for (int i = 0; i < 8; ++i) {
      float mo = __shfl_xor(m[i], off);
      float Zo = __shfl_xor(Z[i], off);
      float mn = fmaxf(m[i], mo);
      Z[i] = Z[i] * __expf(m[i] - mn) + Zo * __expf(mo - mn);
      m[i] = mn;
    }
  }
  int lane = threadIdx.x & 31;
  if ((lane & 15) == 0) {
    size_t basei = (size_t)(b * NH + h) * LL + sb * 16 + ((lane & 16) ? 8 : 0);
#pragma unroll
    for (int i = 0; i < 8; ++i) {
      mcol[basei + i] = m[i];
      rZ[basei + i] = 1.0f / Z[i];
    }
  }
}

// ---------------- attention: catV[b,q,h*64+v] = sum_s P[q,s]*Vp[s,v] ----------------
__global__ void attn_kernel(const __bf16* __restrict__ Qp, const __bf16* __restrict__ Kp,
                            const __bf16* __restrict__ VpT, const float* __restrict__ mcol,
                            const float* __restrict__ rZ, __bf16* __restrict__ catV) {
  int wave = blockIdx.x * (blockDim.x >> 5) + (threadIdx.x >> 5);
  int qb = wave & 127;
  int h  = (wave >> 7) & (NH - 1);
  int b  = wave >> 11;
  size_t head = (size_t)(b * NH + h);
  const __bf16* Qh = Qp  + head * LL * DK;
  const __bf16* Kh = Kp  + head * LL * DK;
  const __bf16* Vh = VpT + head * DK * LL;     // [v][s]
  const float*  mh = mcol + head * LL;
  const float*  zh = rZ   + head * LL;

  v16bf bq0  = load_mat_B(Qh, DK, qb * 16, 0);   // Qp^T fragments, loaded once
  v16bf bq32 = load_mat_B(Qh, DK, qb * 16, 32);

  int lane = threadIdx.x & 31;
  int soff = (lane & 16) ? 8 : 0;

  v8f w0 = {}, w1 = {}, w2 = {}, w3 = {};
  for (int s0 = 0; s0 < LL; s0 += 32) {
    if (s0 + 64 < LL)
      __builtin_prefetch((const char*)(Kh + (size_t)(s0 + 64) * DK) + lane * 128, 0, 1);

    v16bf a00 = load_mat_A(Kh, DK, s0,      0);
    v16bf a01 = load_mat_A(Kh, DK, s0,      32);
    v16bf a10 = load_mat_A(Kh, DK, s0 + 16, 0);
    v16bf a11 = load_mat_A(Kh, DK, s0 + 16, 32);
    v8f st0 = {}, st1 = {};
    st0 = wmma_bf16(a00, bq0,  st0);
    st0 = wmma_bf16(a01, bq32, st0);
    st1 = wmma_bf16(a10, bq0,  st1);
    st1 = wmma_bf16(a11, bq32, st1);

    const float* mp = mh + s0 + soff;
    const float* zp = zh + s0 + soff;
    v16bf pa;
#pragma unroll
    for (int i = 0; i < 8; ++i) {
      float p0 = __expf(st0[i] * SCALE - mp[i])      * zp[i];
      float p1 = __expf(st1[i] * SCALE - mp[16 + i]) * zp[16 + i];
      pa[i]     = (__bf16)p0;
      pa[8 + i] = (__bf16)p1;
    }
    v16bf vb0 = load_mat_B(Vh, LL, 0,  s0);
    v16bf vb1 = load_mat_B(Vh, LL, 16, s0);
    v16bf vb2 = load_mat_B(Vh, LL, 32, s0);
    v16bf vb3 = load_mat_B(Vh, LL, 48, s0);
    w0 = wmma_bf16(pa, vb0, w0);
    w1 = wmma_bf16(pa, vb1, w1);
    w2 = wmma_bf16(pa, vb2, w2);
    w3 = wmma_bf16(pa, vb3, w3);
  }

  int col = lane & 15;
  int rh = (lane & 16) ? 8 : 0;
  v8f ws_[4] = {w0, w1, w2, w3};
#pragma unroll
  for (int nb = 0; nb < 4; ++nb) {
#pragma unroll
    for (int r = 0; r < 8; ++r) {
      int q = qb * 16 + rh + r;
      catV[((size_t)b * LL + q) * (NH * DV) + h * DV + nb * 16 + col] = (__bf16)ws_[nb][r];
    }
  }
}

// ---------------- output projection: out = catV * Wo + bo ----------------
// 8 waves per block share one 64-column Wo panel; TDM-staged like proj.
__global__ void outproj_kernel(const __bf16* __restrict__ catV, const __bf16* __restrict__ WoT,
                               const float* __restrict__ bo, float* __restrict__ out) {
  __shared__ __bf16 sW[2][64 * LDW];
  int wave = blockIdx.x * (blockDim.x >> 5) + (threadIdx.x >> 5);
  int rt = wave & 255;       // 256 row tiles (8 consecutive per block)
  int ct = wave >> 8;        // 16 column tiles of 64, shared within block
  const __bf16* Wc = WoT + (size_t)(ct * 64) * DM;

#if HAVE_TDM
  if (threadIdx.x < 32)
    tdm_stage_panel(Wc, (unsigned)reinterpret_cast<uintptr_t>(&sW[0][0]));
#endif

  v8f acc0 = {}, acc1 = {}, acc2 = {}, acc3 = {};
  for (int c = 0; c < NCH; ++c) {
    const int p = c & 1;
#if HAVE_TDM
    if (threadIdx.x < 32) {
      if (c + 1 < NCH) {
        tdm_stage_panel(Wc + (c + 1) * BK,
                        (unsigned)reinterpret_cast<uintptr_t>(&sW[p ^ 1][0]));
        __builtin_amdgcn_s_wait_tensorcnt(1);
      } else {
        __builtin_amdgcn_s_wait_tensorcnt(0);
      }
    }
#else
    for (int i = threadIdx.x; i < 64 * BK; i += 256) {
      int r = i >> 7, kk = i & (BK - 1);
      sW[p][r * LDW + kk] = Wc[(size_t)r * DM + c * BK + kk];
    }
#endif
    __syncthreads();
    const __bf16* sWp = &sW[p][0];
#pragma unroll
    for (int kk = 0; kk < BK; kk += 32) {
      v16bf a  = load_mat_A(catV, DM, rt * 16, c * BK + kk);
      v16bf b0 = load_mat_B(sWp, LDW, 0,  kk);
      v16bf b1 = load_mat_B(sWp, LDW, 16, kk);
      v16bf b2 = load_mat_B(sWp, LDW, 32, kk);
      v16bf b3 = load_mat_B(sWp, LDW, 48, kk);
      acc0 = wmma_bf16(a, b0, acc0);
      acc1 = wmma_bf16(a, b1, acc1);
      acc2 = wmma_bf16(a, b2, acc2);
      acc3 = wmma_bf16(a, b3, acc3);
    }
    __syncthreads();
  }

  int lane = threadIdx.x & 31;
  int col = lane & 15;
  int rh = (lane & 16) ? 8 : 0;
  v8f accs[4] = {acc0, acc1, acc2, acc3};
#pragma unroll
  for (int nb = 0; nb < 4; ++nb) {
    int mcoli = ct * 64 + nb * 16 + col;
    float bv = bo[mcoli];
#pragma unroll
    for (int r = 0; r < 8; ++r) {
      int row = rt * 16 + rh + r;
      out[(size_t)row * DM + mcoli] = accs[nb][r] + bv;
    }
  }
}

// ---------------- host launch ----------------
extern "C" void kernel_launch(void* const* d_in, const int* in_sizes, int n_in,
                              void* d_out, int out_size, void* d_ws, size_t ws_size,
                              hipStream_t stream) {
  const float* Q  = (const float*)d_in[0];
  const float* K  = (const float*)d_in[1];
  const float* V  = (const float*)d_in[2];
  const float* Wq = (const float*)d_in[3];
  const float* bq = (const float*)d_in[4];
  const float* Wk = (const float*)d_in[5];
  const float* bk = (const float*)d_in[6];
  const float* Wv = (const float*)d_in[7];
  const float* bv = (const float*)d_in[8];
  const float* Wo = (const float*)d_in[9];
  const float* bo = (const float*)d_in[10];
  float* out = (float*)d_out;

  const size_t SZ_X   = (size_t)BB * LL * DM;        // 4,194,304 elems
  const size_t SZ_W   = (size_t)NH * DM * DK;        // 1,048,576 elems
  const size_t SZ_WO  = (size_t)DM * DM;             // 1,048,576 elems
  const size_t SZ_P   = (size_t)BB * NH * LL * DK;   // 4,194,304 elems
  const size_t SZ_ST  = (size_t)BB * NH * LL;        // 65,536 elems

  char* w = (char*)d_ws;
  __bf16* Qb   = (__bf16*)w;  w += SZ_X * 2;
  __bf16* Kb   = (__bf16*)w;  w += SZ_X * 2;
  __bf16* Vb   = (__bf16*)w;  w += SZ_X * 2;
  __bf16* WqT  = (__bf16*)w;  w += SZ_W * 2;
  __bf16* WkT  = (__bf16*)w;  w += SZ_W * 2;
  __bf16* WvT  = (__bf16*)w;  w += SZ_W * 2;
  __bf16* WoT  = (__bf16*)w;  w += SZ_WO * 2;
  __bf16* Qp   = (__bf16*)w;  w += SZ_P * 2;
  __bf16* Kp   = (__bf16*)w;  w += SZ_P * 2;
  __bf16* VpT  = (__bf16*)w;  w += SZ_P * 2;
  float*  mcol = (float*)w;   w += SZ_ST * 4;
  float*  rZ   = (float*)w;   w += SZ_ST * 4;
  __bf16* catV = (__bf16*)w;  w += SZ_X * 2;

  const int TB = 256;
  cvt_bf16_kernel<<<(int)((SZ_X + TB - 1) / TB), TB, 0, stream>>>(Q, Qb, (int)SZ_X);
  cvt_bf16_kernel<<<(int)((SZ_X + TB - 1) / TB), TB, 0, stream>>>(K, Kb, (int)SZ_X);
  cvt_bf16_kernel<<<(int)((SZ_X + TB - 1) / TB), TB, 0, stream>>>(V, Vb, (int)SZ_X);
  transpose_bf16_kernel<<<(int)((SZ_W + TB - 1) / TB), TB, 0, stream>>>(Wq, WqT, DM, DK, (int)SZ_W);
  transpose_bf16_kernel<<<(int)((SZ_W + TB - 1) / TB), TB, 0, stream>>>(Wk, WkT, DM, DK, (int)SZ_W);
  transpose_bf16_kernel<<<(int)((SZ_W + TB - 1) / TB), TB, 0, stream>>>(Wv, WvT, DM, DK, (int)SZ_W);
  transpose_bf16_kernel<<<(int)((SZ_WO + TB - 1) / TB), TB, 0, stream>>>(Wo, WoT, DM, DM, (int)SZ_WO);

  proj_kernel<<<512, TB, 0, stream>>>(Qb, WqT, bq, Qp, 0);
  proj_kernel<<<512, TB, 0, stream>>>(Kb, WkT, bk, Kp, 0);
  proj_kernel<<<512, TB, 0, stream>>>(Vb, WvT, bv, VpT, 1);

  colstats_kernel<<<512, TB, 0, stream>>>(Qp, Kp, mcol, rZ);
  attn_kernel<<<512, TB, 0, stream>>>(Qp, Kp, VpT, mcol, rZ, catV);
  outproj_kernel<<<512, TB, 0, stream>>>(catV, WoT, bo, out);
}
